// AttentionBlockPersistent_6201932775592
// MI455X (gfx1250) — matrix operand order repaired and checked by reading, not compiled, not run on hardware
//
#include <hip/hip_runtime.h>
#include <math.h>

typedef _Float16 f16;
typedef _Float16 __attribute__((ext_vector_type(16))) v16h;
typedef _Float16 __attribute__((ext_vector_type(8)))  v8h;
typedef float    __attribute__((ext_vector_type(8)))  v8f;

#define BATCH 32
#define CH    128
#define NSEQ  1024
#define ACTC  416   // [x:0..127 | pos:128..159 | k:160..287 | v:288..415]

// ---------------- WMMA tile helpers (ISA 7.12.2 layouts, wave32) ----------------
// A-tile 16x32 f16: lane<16 holds row M=lane, K={0..7,16..23}; lane>=16 row M=lane-16, K={8..15,24..31}
__device__ __forceinline__ v16h load_a32(const f16* rowk, int half) {
  union { v16h v; v8h h[2]; } u;
  const int o = half ? 8 : 0;
  u.h[0] = *(const v8h*)(rowk + o);
  u.h[1] = *(const v8h*)(rowk + 16 + o);
  return u.v;
}
// B-tile 32x16 f16: lane holds col N=lane%16, K = (lane<16 ? {0..7,16..23} : {8..15,24..31})
__device__ __forceinline__ v16h load_b32(const f16* colk, int half) {
  union { v16h v; v8h h[2]; } u;
  const int o = half ? 8 : 0;
  u.h[0] = *(const v8h*)(colk + o);
  u.h[1] = *(const v8h*)(colk + 16 + o);
  return u.v;
}
__device__ __forceinline__ v8f wmma16(v16h a, v16h b, v8f c) {
  return __builtin_amdgcn_wmma_f32_16x16x32_f16(false, a, false, b, (short)0, c, false, false);
}
__device__ __forceinline__ v8f vzero8() {
  v8f z = {0.f,0.f,0.f,0.f,0.f,0.f,0.f,0.f};
  return z;
}

// ---------------- positional encoding (faithful raw-reinterpret of [32,16] table) ----
__device__ __forceinline__ float posval(int e, int s) {
  int fi = e * 32 + s;          // flat index into row-major [32][16] table
  int p = fi >> 4, j = fi & 15;
  if (p == 0) return 0.f;
  float ang = (float)p * __powf(10000.f, -(float)(2 * (j >> 1)) / 16.f);
  return (j & 1) ? __cosf(ang) : __sinf(ang);
}

// ---------------- prep kernels ----------------
__global__ void k_f32_to_f16(const float* __restrict__ s, f16* __restrict__ d, int n) {
  int i = blockIdx.x * blockDim.x + threadIdx.x;
  if (i < n) d[i] = (f16)s[i];
}

__global__ void k_build_actT(const float* __restrict__ x, const float* __restrict__ k,
                             const float* __restrict__ v, f16* __restrict__ actT) {
  long i = (long)blockIdx.x * blockDim.x + threadIdx.x;
  const long total = (long)BATCH * NSEQ * ACTC;
  if (i >= total) return;
  int c = (int)(i % ACTC);
  int n = (int)((i / ACTC) % NSEQ);
  int b = (int)(i / ((long)ACTC * NSEQ));
  float val;
  if (c < 128) {
    val = x[((long)b * CH + c) * NSEQ + n];
  } else if (c < 160) {
    int e = c - 128;
    val = (e < 16) ? posval(e, n >> 5)        // xe: depends on row
                   : posval(e - 16, n & 31);  // ye: depends on col
  } else if (c < 288) {
    val = k[((long)b * CH + (c - 160)) * NSEQ + n];
  } else {
    val = v[((long)b * CH + (c - 288)) * NSEQ + n];
  }
  actT[i] = (f16)val;   // layout (b*N + n)*ACTC + c  == flat i
}

// ------- q = relu(Wq @ x0 + bq), transposed qT[b][n][128]; 2 o-tiles/wave share B -------
__global__ void __launch_bounds__(256) k_qgemm(const f16* __restrict__ Wq, const float* __restrict__ bq,
                                               const f16* __restrict__ actT, f16* __restrict__ qT) {
  int wid  = (blockIdx.x * blockDim.x + threadIdx.x) >> 5;
  int lane = threadIdx.x & 31, half = lane >> 4, lp = lane & 15;
  int b = wid >> 8, r = wid & 255, o0 = ((r >> 6) & 3) * 32, n0 = (r & 63) * 16;
  v8f acc0 = vzero8(), acc1 = vzero8();
  const f16* brow = actT + (size_t)(b * NSEQ + n0 + lp) * ACTC;
  for (int cb = 0; cb < 160; cb += 32) {
    v16h bt = load_b32(brow + cb, half);
    v16h a0 = load_a32(Wq + (o0 + lp) * 160 + cb, half);
    v16h a1 = load_a32(Wq + (o0 + 16 + lp) * 160 + cb, half);
    acc0 = wmma16(a0, bt, acc0);
    acc1 = wmma16(a1, bt, acc1);
  }
  for (int t = 0; t < 2; ++t) {
    v8f acc = t ? acc1 : acc0;
    for (int i = 0; i < 8; ++i) {
      int o = o0 + t * 16 + i + (half ? 8 : 0);
      float q = acc[i] + bq[o];
      qT[(size_t)(b * NSEQ + n0 + lp) * CH + o] = (f16)(q > 0.f ? q : 0.f);
    }
  }
}

// ---- fused highway gates: dst = sigmoid(Wf@in+bf)*src + relu(Wc@in+bc) ----
// 2 o-tiles per wave share each activation B-tile -> 4 WMMA per B load
__global__ void __launch_bounds__(256) k_gates(const f16* __restrict__ Wf, const f16* __restrict__ Wc,
                                               const float* __restrict__ bf, const float* __restrict__ bc,
                                               const f16* __restrict__ actT, const float* __restrict__ src,
                                               float* __restrict__ dst, int vmode) {
  int wid  = (blockIdx.x * blockDim.x + threadIdx.x) >> 5;
  int lane = threadIdx.x & 31, half = lane >> 4, lp = lane & 15;
  int b = wid >> 8, r = wid & 255, o0 = ((r >> 6) & 3) * 32, n0 = (r & 63) * 16;
  v8f accf0 = vzero8(), accc0 = vzero8(), accf1 = vzero8(), accc1 = vzero8();
  const f16* brow = actT + (size_t)(b * NSEQ + n0 + lp) * ACTC;
  for (int cb = 0; cb < 288; cb += 32) {
    int ach = (vmode && cb >= 160) ? cb + 128 : cb;  // x0v skips the k channel block
    v16h bt  = load_b32(brow + ach, half);
    v16h aF0 = load_a32(Wf + (o0 + lp) * 288 + cb, half);
    v16h aC0 = load_a32(Wc + (o0 + lp) * 288 + cb, half);
    v16h aF1 = load_a32(Wf + (o0 + 16 + lp) * 288 + cb, half);
    v16h aC1 = load_a32(Wc + (o0 + 16 + lp) * 288 + cb, half);
    accf0 = wmma16(aF0, bt, accf0);
    accc0 = wmma16(aC0, bt, accc0);
    accf1 = wmma16(aF1, bt, accf1);
    accc1 = wmma16(aC1, bt, accc1);
  }
  for (int t = 0; t < 2; ++t) {
    v8f af = t ? accf1 : accf0;
    v8f ac = t ? accc1 : accc0;
    for (int i = 0; i < 8; ++i) {
      int o = o0 + t * 16 + i + (half ? 8 : 0);
      int n = n0 + lp;
      float f = 1.f / (1.f + __expf(-(af[i] + bf[o])));
      float c = ac[i] + bc[o];
      c = c > 0.f ? c : 0.f;
      size_t idx = (size_t)b * CH * NSEQ + (size_t)o * NSEQ + n;
      dst[idx] = f * src[idx] + c;
    }
  }
}

// ---------------- flash causal attention: one wave per (batch, 16-row query tile) -----
__global__ void __launch_bounds__(128) k_attn(const f16* __restrict__ qT, const f16* __restrict__ actT,
                                              const f16* __restrict__ vf, float* __restrict__ out) {
  __shared__ f16 pst[4][16 * 32];
  int wib  = threadIdx.x >> 5;
  int wid  = blockIdx.x * 4 + wib;
  int lane = threadIdx.x & 31, half = lane >> 4, lp = lane & 15;
  int b = wid >> 6, n0 = (wid & 63) * 16;

  v16h aq[4];
  for (int kk = 0; kk < 4; ++kk)
    aq[kk] = load_a32(qT + (size_t)(b * NSEQ + n0 + lp) * CH + kk * 32, half);

  float mrun[8], lrun[8];
  v8f oacc[8];
  for (int i = 0; i < 8; ++i) { mrun[i] = -1e30f; lrun[i] = 0.f; }
  for (int vt = 0; vt < 8; ++vt) oacc[vt] = vzero8();

  const float qs = 0.088388347648318447f; // 1/sqrt(128)
  f16* lds = pst[wib];
  const int mbe = (n0 + 15) >> 5;         // inclusive diagonal block

  for (int mb = 0; mb <= mbe; ++mb) {
    int m0 = mb * 32;
    // speculative prefetch of next m-block K rows + V span (global_prefetch_b8,
    // no LOADcnt cost; overlaps L2 latency with the softmax below)
    if (mb < mbe) {
      __builtin_prefetch(actT + (size_t)(b * NSEQ + m0 + 32 + lane) * ACTC + 160, 0, 3);
      __builtin_prefetch(vf + (size_t)(b * CH + lane * 4) * NSEQ + m0 + 32, 0, 3);
    }
    v8f p0 = vzero8(), p1 = vzero8();
    for (int kk = 0; kk < 4; ++kk) {      // S = q^T k over head-dim 128
      v16h b0 = load_b32(actT + (size_t)(b * NSEQ + m0 + lp) * ACTC + 160 + kk * 32, half);
      v16h b1 = load_b32(actT + (size_t)(b * NSEQ + m0 + 16 + lp) * ACTC + 160 + kk * 32, half);
      p0 = wmma16(aq[kk], b0, p0);
      p1 = wmma16(aq[kk], b1, p1);
    }
    int c0 = m0 + lp, c1 = m0 + 16 + lp;
    for (int i = 0; i < 8; ++i) {
      int n = n0 + i + (half ? 8 : 0);
      float x0 = p0[i] * qs, x1 = p1[i] * qs;
      if (c0 > n) x0 = -1e30f;            // causal mask -> exact zero prob, matches post-mask
      if (c1 > n) x1 = -1e30f;
      float mx = fmaxf(x0, x1);
      for (int off = 8; off; off >>= 1) mx = fmaxf(mx, __shfl_xor(mx, off, 32));
      float nm = fmaxf(mrun[i], mx);
      float sc = __expf(mrun[i] - nm);
      mrun[i] = nm;
      x0 = __expf(x0 - nm);
      x1 = __expf(x1 - nm);
      float rs = x0 + x1;
      for (int off = 8; off; off >>= 1) rs += __shfl_xor(rs, off, 32);
      lrun[i] = lrun[i] * sc + rs;
      for (int vt = 0; vt < 8; ++vt) oacc[vt][i] *= sc;
      int row = i + (half ? 8 : 0);       // D-layout -> LDS [row][col] restage
      lds[row * 32 + lp]      = (f16)x0;
      lds[row * 32 + 16 + lp] = (f16)x1;
    }
    __builtin_amdgcn_wave_barrier();
    asm volatile("s_wait_dscnt 0" ::: "memory");
    v16h ap = load_a32(lds + lp * 32, half);   // P in A-layout (K = 32 keys)
    for (int vt = 0; vt < 8; ++vt) {           // O += P @ V^T
      v16h bv = load_b32(vf + (size_t)(b * CH + vt * 16 + lp) * NSEQ + m0, half);
      oacc[vt] = wmma16(ap, bv, oacc[vt]);
    }
    __builtin_amdgcn_wave_barrier();
  }
  for (int vt = 0; vt < 8; ++vt)
    for (int i = 0; i < 8; ++i) {
      int n = n0 + i + (half ? 8 : 0);
      int vch = vt * 16 + lp;
      out[(size_t)b * CH * NSEQ + (size_t)vch * NSEQ + n] = oacc[vt][i] / lrun[i];
    }
}

// ---------------- launch ----------------
extern "C" void kernel_launch(void* const* d_in, const int* in_sizes, int n_in,
                              void* d_out, int out_size, void* d_ws, size_t ws_size,
                              hipStream_t stream) {
  const float* x    = (const float*)d_in[0];
  const float* k    = (const float*)d_in[1];
  const float* v    = (const float*)d_in[2];
  const float* W_q  = (const float*)d_in[3];
  const float* b_q  = (const float*)d_in[4];
  const float* W_fk = (const float*)d_in[5];
  const float* b_fk = (const float*)d_in[6];
  const float* W_ck = (const float*)d_in[7];
  const float* b_ck = (const float*)d_in[8];
  const float* W_fv = (const float*)d_in[9];
  const float* b_fv = (const float*)d_in[10];
  const float* W_cv = (const float*)d_in[11];
  const float* b_cv = (const float*)d_in[12];
  float* out = (float*)d_out;

  char* ws = (char*)d_ws;
  size_t off = 0;
  f16* actT = (f16*)(ws + off); off += (size_t)BATCH * NSEQ * ACTC * 2;  // 27.3 MB
  f16* qT   = (f16*)(ws + off); off += (size_t)BATCH * NSEQ * CH * 2;    //  8.4 MB
  f16* vf   = (f16*)(ws + off); off += (size_t)BATCH * CH * NSEQ * 2;    //  8.4 MB
  f16* Wq_h  = (f16*)(ws + off); off += (size_t)128 * 160 * 2;
  f16* Wfk_h = (f16*)(ws + off); off += (size_t)128 * 288 * 2;
  f16* Wck_h = (f16*)(ws + off); off += (size_t)128 * 288 * 2;
  f16* Wfv_h = (f16*)(ws + off); off += (size_t)128 * 288 * 2;
  f16* Wcv_h = (f16*)(ws + off); off += (size_t)128 * 288 * 2;

  const int TB = 256;
  k_f32_to_f16<<<(128 * 160 + TB - 1) / TB, TB, 0, stream>>>(W_q,  Wq_h,  128 * 160);
  k_f32_to_f16<<<(128 * 288 + TB - 1) / TB, TB, 0, stream>>>(W_fk, Wfk_h, 128 * 288);
  k_f32_to_f16<<<(128 * 288 + TB - 1) / TB, TB, 0, stream>>>(W_ck, Wck_h, 128 * 288);
  k_f32_to_f16<<<(128 * 288 + TB - 1) / TB, TB, 0, stream>>>(W_fv, Wfv_h, 128 * 288);
  k_f32_to_f16<<<(128 * 288 + TB - 1) / TB, TB, 0, stream>>>(W_cv, Wcv_h, 128 * 288);

  long actn = (long)BATCH * NSEQ * ACTC;
  k_build_actT<<<(unsigned)((actn + TB - 1) / TB), TB, 0, stream>>>(x, k, v, actT);
  k_f32_to_f16<<<(BATCH * CH * NSEQ + TB - 1) / TB, TB, 0, stream>>>(v, vf, BATCH * CH * NSEQ);

  // 32 batches * 4 o-pair tiles * 64 n-tiles = 8192 waves -> 1024 blocks of 256
  k_qgemm<<<1024, 256, 0, stream>>>(Wq_h, b_q, actT, qT);
  k_gates<<<1024, 256, 0, stream>>>(Wfk_h, Wck_h, b_fk, b_ck, actT, k,
                                    out + (size_t)BATCH * CH * NSEQ, 0);
  k_gates<<<1024, 256, 0, stream>>>(Wfv_h, Wcv_h, b_fv, b_cv, actT, v,
                                    out + 2 * (size_t)BATCH * CH * NSEQ, 1);

  // 32 batches * 64 query tiles = 2048 waves -> 512 blocks of 128 (4 waves each)
  k_attn<<<512, 128, 0, stream>>>(qT, actT, vf, out);
}
